// MultiHeadSelfAttention_2130303779352
// MI455X (gfx1250) — matrix operand order repaired and checked by reading, not compiled, not run on hardware
//
#include <hip/hip_runtime.h>
#include <hip/hip_bf16.h>

// ---------------------------------------------------------------------------
// MI455X (gfx1250) multi-head self-attention:
//   bf16 WMMA GEMMs (double-buffered, TDM-staged A tiles) + flash attention.
// ---------------------------------------------------------------------------

typedef __attribute__((ext_vector_type(16))) __bf16       v16bf;
typedef __attribute__((ext_vector_type(8)))  float        v8f;
typedef __attribute__((ext_vector_type(4)))  unsigned int su4;
typedef __attribute__((ext_vector_type(8)))  unsigned int su8;

#define BATCH  4
#define SEQ    2048
#define CH     1024
#define HEADS  16
#define HDIM   64
#define C3     3072
#define MROWS  (BATCH * SEQ)   // 8192
#define TILEK  32
#define LDSA   (128 * TILEK)   // elements per A/B buffer

union Frag {                   // one wave32 16-bf16 fragment (8 VGPRs)
    v16bf v;
    uint4 u[2];
};

__device__ inline v8f wmma_bf16(const v16bf a, const v16bf b, const v8f c) {
    // D = A(16x32 bf16) * B(32x16 bf16) + C(16x16 f32)
    return __builtin_amdgcn_wmma_f32_16x16x32_bf16(
        /*neg_a=*/false, a, /*neg_b=*/false, b,
        /*c_mod=*/(short)0, c, /*reuse_a=*/false, /*reuse_b=*/false);
}

// Issue a TDM 2D tile load: A[m0 + 0..127][kk + 0..31] (row stride K) -> LDS,
// landing contiguously as As[row*32 + col].
__device__ inline void tdm_load_A(const __bf16* A, int K, int m0, int kk,
                                  unsigned int lds_addr) {
    unsigned long long ga =
        (unsigned long long)(size_t)A + ((size_t)m0 * K + kk) * 2ull;
    su4 g0;
    g0[0] = 1u;                                        // count=1, user mode
    g0[1] = lds_addr;                                  // lds_addr
    g0[2] = (unsigned int)ga;                          // global_addr[31:0]
    g0[3] = (unsigned int)(ga >> 32) | (2u << 30);     // ga[56:32] | type=2
    su8 g1;
    g1[0] = 1u << 16;                                  // data_size = 2 bytes
    g1[1] = ((unsigned int)K & 0xFFFFu) << 16;         // tensor_dim0[15:0]
    g1[2] = ((unsigned int)K >> 16) |
            (((unsigned int)MROWS & 0xFFFFu) << 16);   // td0 hi | td1 lo
    g1[3] = ((unsigned int)MROWS >> 16) | (32u << 16); // td1 hi | tile_dim0
    g1[4] = 128u;                                      // tile_dim1 (tile2=0)
    g1[5] = (unsigned int)K;                           // tensor_dim0_stride lo
    g1[6] = 0u;
    g1[7] = 0u;
    asm volatile("tensor_load_to_lds %0, %1" :: "s"(g0), "s"(g1) : "memory");
}

// ---------------------------------------------------------------------------
// f32 -> bf16 conversion, 8 elements / thread
// ---------------------------------------------------------------------------
__global__ __launch_bounds__(256) void k_cvt_bf16(const float* __restrict__ src,
                                                  __bf16* __restrict__ dst, int n) {
    int i = (blockIdx.x * 256 + threadIdx.x) * 8;
    if (i + 8 <= n) {
        float4 a = *(const float4*)&src[i];
        float4 b = *(const float4*)&src[i + 4];
        union { __bf16 h[8]; uint4 u; } t;
        t.h[0] = (__bf16)a.x; t.h[1] = (__bf16)a.y;
        t.h[2] = (__bf16)a.z; t.h[3] = (__bf16)a.w;
        t.h[4] = (__bf16)b.x; t.h[5] = (__bf16)b.y;
        t.h[6] = (__bf16)b.z; t.h[7] = (__bf16)b.w;
        *(uint4*)&dst[i] = t.u;
    }
}

// ---------------------------------------------------------------------------
// Double-buffered WMMA GEMM mainloop: 256 threads (8 waves), block tile
// 128(M) x 128(N), K stepped by 32.  Waves arranged 4(M) x 2(N): each wave
// owns 32 x 64 (2 A-frags x 4 B-frags -> 8 WMMA / k-step).
//
// Pipeline per iteration i:
//   1. wave0 issues TDM for A tile i+1 into the spare buffer; all threads
//      issue global loads of W tile i+1 (in flight during compute)
//   2. compute 8 WMMAs from buffer i (guaranteed by previous barrier)
//   3. scatter arrived W data transposed into the spare Bs buffer
//   4. wave0 drains TENSORcnt; __syncthreads() drains DS stores
// ---------------------------------------------------------------------------
__device__ inline void gemm_tile(const __bf16* __restrict__ A,
                                 const __bf16* __restrict__ W,
                                 int K, int N, int m0, int n0,
                                 __bf16* As, __bf16* Bs, v8f acc[2][4]) {
    const int tid  = threadIdx.x;
    const int wave = tid >> 5;
    const int lane = tid & 31;
    const int ml   = lane & 15;
    const int hi   = lane >> 4;
    const int wr   = wave >> 1;      // 0..3 : 32-row band
    const int wc   = wave & 1;       // 0..1 : 64-col band

    // LDS byte addresses: flat VA low 32 bits == LDS address (ISA 10.2)
    const unsigned int lds_a0 = (unsigned int)(size_t)(const void*)As;
    const unsigned int lds_a1 = (unsigned int)(size_t)(const void*)(As + LDSA);

    const int krow = tid >> 4;       // 0..15 (second chunk: +16)
    const int n8   = (tid & 15) * 8;

    // ---- prologue: stage tile 0 into buffer 0 ----
    if (wave == 0) tdm_load_A(A, K, m0, 0, lds_a0);
    {
        uint4 wv0 = *(const uint4*)&W[(size_t)krow * N + n0 + n8];
        uint4 wv1 = *(const uint4*)&W[(size_t)(krow + 16) * N + n0 + n8];
        const __bf16* p0 = (const __bf16*)&wv0;
        const __bf16* p1 = (const __bf16*)&wv1;
#pragma unroll
        for (int j = 0; j < 8; ++j) {
            Bs[(n8 + j) * 32 + krow]      = p0[j];
            Bs[(n8 + j) * 32 + krow + 16] = p1[j];
        }
    }
    if (wave == 0) __builtin_amdgcn_s_wait_tensorcnt(0);
    __syncthreads();

    int buf = 0;
    for (int kk = 0; kk < K; kk += TILEK) {
        const int  nxt  = kk + TILEK;
        const bool more = nxt < K;                 // uniform
        // ---- 1. kick off next tile's data movement ----
        uint4 wv0, wv1;
        if (more) {
            if (wave == 0) tdm_load_A(A, K, m0, nxt, buf ? lds_a0 : lds_a1);
            wv0 = *(const uint4*)&W[(size_t)(nxt + krow) * N + n0 + n8];
            wv1 = *(const uint4*)&W[(size_t)(nxt + krow + 16) * N + n0 + n8];
        }
        // ---- 2. compute on current buffer ----
        const __bf16* Asb = As + buf * LDSA;
        const __bf16* Bsb = Bs + buf * LDSA;
        Frag af[2];
#pragma unroll
        for (int ra = 0; ra < 2; ++ra) {
            int row = wr * 32 + ra * 16 + ml;
            af[ra].u[0] = *(const uint4*)&Asb[row * 32 + hi * 8];
            af[ra].u[1] = *(const uint4*)&Asb[row * 32 + hi * 8 + 16];
        }
#pragma unroll
        for (int j = 0; j < 4; ++j) {
            Frag bf;
            int n = wc * 64 + j * 16 + ml;
            bf.u[0] = *(const uint4*)&Bsb[n * 32 + hi * 8];
            bf.u[1] = *(const uint4*)&Bsb[n * 32 + hi * 8 + 16];
#pragma unroll
            for (int ra = 0; ra < 2; ++ra)
                acc[ra][j] = wmma_bf16(af[ra].v, bf.v, acc[ra][j]);
        }
        // ---- 3. land next W tile (transposed) into spare buffer ----
        if (more) {
            __bf16* Bsn = Bs + (buf ^ 1) * LDSA;
            const __bf16* p0 = (const __bf16*)&wv0;
            const __bf16* p1 = (const __bf16*)&wv1;
#pragma unroll
            for (int j = 0; j < 8; ++j) {
                Bsn[(n8 + j) * 32 + krow]      = p0[j];
                Bsn[(n8 + j) * 32 + krow + 16] = p1[j];
            }
        }
        // ---- 4. make next buffer visible ----
        if (wave == 0) __builtin_amdgcn_s_wait_tensorcnt(0);
        __syncthreads();
        buf ^= 1;
    }
}

// ---------------------------------------------------------------------------
// QKV projection: qkv = x @ qkv_w + qkv_b, scattered to
//   Q,K : [B,H,T,D] bf16       V : [B,H,D,T] bf16 (transposed for PV WMMA)
// ---------------------------------------------------------------------------
__global__ __launch_bounds__(256) void k_gemm_qkv(const __bf16* __restrict__ xb,
                                                  const __bf16* __restrict__ wb,
                                                  const float*  __restrict__ bias,
                                                  __bf16* __restrict__ qo,
                                                  __bf16* __restrict__ ko,
                                                  __bf16* __restrict__ vto) {
    __shared__ __bf16 As[2 * LDSA];
    __shared__ __bf16 Bs[2 * LDSA];
    v8f acc[2][4];
#pragma unroll
    for (int ra = 0; ra < 2; ++ra)
#pragma unroll
        for (int j = 0; j < 4; ++j)
#pragma unroll
            for (int e = 0; e < 8; ++e) acc[ra][j][e] = 0.0f;

    const int m0 = blockIdx.x * 128;
    const int n0 = blockIdx.y * 128;
    gemm_tile(xb, wb, CH, C3, m0, n0, As, Bs, acc);

    const int tid  = threadIdx.x;
    const int wave = tid >> 5;
    const int lane = tid & 31;
    const int ml   = lane & 15;
    const int hi   = lane >> 4;
    const int wr   = wave >> 1;
    const int wc   = wave & 1;
    // whole 128-wide N tile is in one qkv section since 128 | 1024
    const int s = n0 >> 10;               // 0=q 1=k 2=v
#pragma unroll
    for (int ra = 0; ra < 2; ++ra) {
#pragma unroll
        for (int j = 0; j < 4; ++j) {
            int nn = n0 + wc * 64 + j * 16 + ml;
            int h  = (nn & 1023) >> 6;
            int d  = nn & 63;
            float bv = bias[nn];
#pragma unroll
            for (int r = 0; r < 8; ++r) {
                int m = m0 + wr * 32 + ra * 16 + r + 8 * hi;
                int b = m >> 11;              // / SEQ
                int t = m & (SEQ - 1);
                float val = acc[ra][j][r] + bv;
                size_t bh = (size_t)(b * HEADS + h);
                if (s == 0)      qo[(bh * SEQ + t) * HDIM + d] = (__bf16)val;
                else if (s == 1) ko[(bh * SEQ + t) * HDIM + d] = (__bf16)val;
                else             vto[(bh * HDIM + d) * SEQ + t] = (__bf16)val;
            }
        }
    }
}

// ---------------------------------------------------------------------------
// Flash attention: grid = B*H*(T/64) blocks of 128 threads (4 waves).
// Each wave owns 16 query rows; streams keys 32 at a time.
// ---------------------------------------------------------------------------
__global__ __launch_bounds__(128) void k_attn(const __bf16* __restrict__ q,
                                              const __bf16* __restrict__ k,
                                              const __bf16* __restrict__ vt,
                                              __bf16* __restrict__ ao) {
    __shared__ __bf16 P[4 * 16 * 32];     // per-wave 16x32 bf16 P tile

    const int tid  = threadIdx.x;
    const int wave = tid >> 5;
    const int lane = tid & 31;
    const int ml   = lane & 15;
    const int hi   = lane >> 4;

    const int bh = blockIdx.x >> 5;       // 0..63  (T/64 == 32 tiles)
    const int t0 = (blockIdx.x & 31) * 64 + wave * 16;
    const int b  = bh >> 4;
    const int h  = bh & 15;

    const __bf16* Q  = q  + ((size_t)bh * SEQ + t0) * HDIM;
    const __bf16* Kb = k  + (size_t)bh * SEQ * HDIM;
    const __bf16* Vt = vt + (size_t)bh * HDIM * SEQ;
    __bf16* Pw = &P[wave * 512];

    // Q fragments (rows fixed for whole loop): 2 k-steps over d
    Frag qa[2];
#pragma unroll
    for (int kk = 0; kk < 2; ++kk) {
        qa[kk].u[0] = *(const uint4*)&Q[ml * HDIM + kk * 32 + hi * 8];
        qa[kk].u[1] = *(const uint4*)&Q[ml * HDIM + kk * 32 + hi * 8 + 16];
    }

    v8f o[4];
#pragma unroll
    for (int j = 0; j < 4; ++j)
#pragma unroll
        for (int e = 0; e < 8; ++e) o[j][e] = 0.0f;
    float mi[8], li[8];
#pragma unroll
    for (int r = 0; r < 8; ++r) { mi[r] = -1e30f; li[r] = 0.0f; }

    for (int s0 = 0; s0 < SEQ; s0 += 32) {
        // ---- S = Q * K^T for 32 keys (two 16x16 tiles) ----
        v8f sa, sb;
#pragma unroll
        for (int e = 0; e < 8; ++e) { sa[e] = 0.0f; sb[e] = 0.0f; }
        const __bf16* Kp = Kb + (size_t)s0 * HDIM;
#pragma unroll
        for (int kk = 0; kk < 2; ++kk) {
            Frag kf;                       // keys s0..s0+15, B-frag table order in d
            kf.u[0] = *(const uint4*)&Kp[ml * HDIM + kk * 32 + hi * 8];
            kf.u[1] = *(const uint4*)&Kp[ml * HDIM + kk * 32 + hi * 8 + 16];
            sa = wmma_bf16(qa[kk].v, kf.v, sa);
            kf.u[0] = *(const uint4*)&Kp[(16 + ml) * HDIM + kk * 32 + hi * 8];
            kf.u[1] = *(const uint4*)&Kp[(16 + ml) * HDIM + kk * 32 + hi * 8 + 16];
            sb = wmma_bf16(qa[kk].v, kf.v, sb);
        }
        // ---- online softmax over this key block ----
#pragma unroll
        for (int r = 0; r < 8; ++r) {
            float x0 = sa[r] * 0.125f;     // 1/sqrt(64)
            float x1 = sb[r] * 0.125f;
            float mx = fmaxf(x0, x1);
            mx = fmaxf(mx, __shfl_xor(mx, 1, 32));
            mx = fmaxf(mx, __shfl_xor(mx, 2, 32));
            mx = fmaxf(mx, __shfl_xor(mx, 4, 32));
            mx = fmaxf(mx, __shfl_xor(mx, 8, 32));
            float mnew  = fmaxf(mi[r], mx);
            float alpha = __expf(mi[r] - mnew);
            float p0 = __expf(x0 - mnew);
            float p1 = __expf(x1 - mnew);
            float rs = p0 + p1;
            rs += __shfl_xor(rs, 1, 32);
            rs += __shfl_xor(rs, 2, 32);
            rs += __shfl_xor(rs, 4, 32);
            rs += __shfl_xor(rs, 8, 32);
            li[r] = li[r] * alpha + rs;
            mi[r] = mnew;
            int row = r + 8 * hi;          // C-layout row of this element
            Pw[row * 32 + ml]      = (__bf16)p0;
            Pw[row * 32 + 16 + ml] = (__bf16)p1;
#pragma unroll
            for (int j = 0; j < 4; ++j) o[j][r] *= alpha;
        }
        // ---- re-read P as A fragment (same-wave LDS ops are in-order) ----
        Frag pf;
        pf.u[0] = *(const uint4*)&Pw[ml * 32 + hi * 8];
        pf.u[1] = *(const uint4*)&Pw[ml * 32 + hi * 8 + 16];
        // ---- O += P(16x32) * V(32x64) using transposed V ----
#pragma unroll
        for (int j = 0; j < 4; ++j) {
            Frag vf;
            vf.u[0] = *(const uint4*)&Vt[(size_t)(j * 16 + ml) * SEQ + s0 + hi * 8];
            vf.u[1] = *(const uint4*)&Vt[(size_t)(j * 16 + ml) * SEQ + s0 + hi * 8 + 16];
            o[j] = wmma_bf16(pf.v, vf.v, o[j]);
        }
    }
    // ---- normalize, write [B,T,C] bf16 for projection GEMM ----
#pragma unroll
    for (int r = 0; r < 8; ++r) {
        float inv = 1.0f / li[r];
        int t = t0 + r + 8 * hi;
#pragma unroll
        for (int j = 0; j < 4; ++j)
            ao[((size_t)(b * SEQ + t)) * CH + h * 64 + j * 16 + ml] =
                (__bf16)(o[j][r] * inv);
    }
}

// ---------------------------------------------------------------------------
// Output projection: out = attn @ proj_w + proj_b  (f32 result)
// ---------------------------------------------------------------------------
__global__ __launch_bounds__(256) void k_gemm_proj(const __bf16* __restrict__ ab,
                                                   const __bf16* __restrict__ wb,
                                                   const float*  __restrict__ bias,
                                                   float* __restrict__ out) {
    __shared__ __bf16 As[2 * LDSA];
    __shared__ __bf16 Bs[2 * LDSA];
    v8f acc[2][4];
#pragma unroll
    for (int ra = 0; ra < 2; ++ra)
#pragma unroll
        for (int j = 0; j < 4; ++j)
#pragma unroll
            for (int e = 0; e < 8; ++e) acc[ra][j][e] = 0.0f;

    const int m0 = blockIdx.x * 128;
    const int n0 = blockIdx.y * 128;
    gemm_tile(ab, wb, CH, CH, m0, n0, As, Bs, acc);

    const int tid  = threadIdx.x;
    const int wave = tid >> 5;
    const int lane = tid & 31;
    const int ml   = lane & 15;
    const int hi   = lane >> 4;
    const int wr   = wave >> 1;
    const int wc   = wave & 1;
#pragma unroll
    for (int ra = 0; ra < 2; ++ra) {
#pragma unroll
        for (int j = 0; j < 4; ++j) {
            int nn = n0 + wc * 64 + j * 16 + ml;
            float bv = bias[nn];
#pragma unroll
            for (int r = 0; r < 8; ++r) {
                int m = m0 + wr * 32 + ra * 16 + r + 8 * hi;
                out[(size_t)m * CH + nn] = acc[ra][j][r] + bv;
            }
        }
    }
}

// ---------------------------------------------------------------------------
// Host launcher
// ---------------------------------------------------------------------------
extern "C" void kernel_launch(void* const* d_in, const int* in_sizes, int n_in,
                              void* d_out, int out_size, void* d_ws, size_t ws_size,
                              hipStream_t stream) {
    const float* x      = (const float*)d_in[0];
    const float* qkv_w  = (const float*)d_in[1];
    const float* qkv_b  = (const float*)d_in[2];
    const float* proj_w = (const float*)d_in[3];
    const float* proj_b = (const float*)d_in[4];

    char* ws = (char*)d_ws;
    constexpr size_t SZ_XB    = (size_t)MROWS * CH * 2;          // 16 MB
    constexpr size_t SZ_WQKV  = (size_t)CH * C3 * 2;             //  6 MB
    constexpr size_t SZ_WPROJ = (size_t)CH * CH * 2;             //  2 MB
    constexpr size_t SZ_HEADT = (size_t)BATCH * HEADS * SEQ * HDIM * 2; // 16 MB

    __bf16* xb   = (__bf16*)(ws);
    __bf16* wqkv = (__bf16*)(ws + SZ_XB);
    __bf16* wprj = (__bf16*)(ws + SZ_XB + SZ_WQKV);
    __bf16* qb   = (__bf16*)(ws + SZ_XB + SZ_WQKV + SZ_WPROJ);
    __bf16* kb   = (__bf16*)(ws + SZ_XB + SZ_WQKV + SZ_WPROJ + SZ_HEADT);
    __bf16* vtb  = (__bf16*)(ws + SZ_XB + SZ_WQKV + SZ_WPROJ + 2 * SZ_HEADT);
    __bf16* aob  = (__bf16*)(ws + SZ_XB + SZ_WQKV + SZ_WPROJ + 3 * SZ_HEADT);

    // 1) bf16 conversions (2048 elems / block)
    k_cvt_bf16<<<(MROWS * CH) / 2048, 256, 0, stream>>>(x, xb, MROWS * CH);
    k_cvt_bf16<<<(CH * C3)   / 2048, 256, 0, stream>>>(qkv_w, wqkv, CH * C3);
    k_cvt_bf16<<<(CH * CH)   / 2048, 256, 0, stream>>>(proj_w, wprj, CH * CH);

    // 2) QKV projection GEMM  (8192 x 3072 x 1024), 128x128 tiles
    k_gemm_qkv<<<dim3(MROWS / 128, C3 / 128), 256, 0, stream>>>(
        xb, wqkv, qkv_b, qb, kb, vtb);

    // 3) flash attention  (B*H*(T/64) = 2048 blocks)
    k_attn<<<BATCH * HEADS * (SEQ / 64), 128, 0, stream>>>(qb, kb, vtb, aob);

    // 4) output projection GEMM  (8192 x 1024 x 1024), f32 out
    k_gemm_proj<<<dim3(MROWS / 128, CH / 128), 256, 0, stream>>>(
        aob, wprj, proj_b, (float*)d_out);
}
